// MemoryAttention_17334488006854
// MI455X (gfx1250) — matrix operand order repaired and checked by reading, not compiled, not run on hardware
//
#include <hip/hip_runtime.h>

typedef __attribute__((ext_vector_type(16))) _Float16 v16h;
typedef __attribute__((ext_vector_type(8)))  _Float16 v8h;
typedef __attribute__((ext_vector_type(8)))  float    v8f;
typedef __attribute__((ext_vector_type(4)))  unsigned int v4u;
typedef __attribute__((ext_vector_type(8)))  int      v8i;
typedef __attribute__((ext_vector_type(4)))  int      v4i;

#define D_MODEL 1024
#define SEQ     2048
#define NB      2
#define NH      16
#define HD      64
#define NEG_FILL (-1.0e11f)

// ---------------------------------------------------------------------------
// WMMA wrapper: D = A(16x32 f16) * B(32x16 f16) + C(16x16 f32)
// ---------------------------------------------------------------------------
static __device__ __forceinline__ v8f wmma16(v16h a, v16h b, v8f c) {
  return __builtin_amdgcn_wmma_f32_16x16x32_f16(
      false, a, false, b, (short)0, c, false, false);
}

// ---------------------------------------------------------------------------
// Tensor Data Mover: 2-D tile load (f16 elements) into LDS.
// D# per CDNA5 ISA:
//  group0: [1:0]=count=1, [63:32]=lds_addr, [120:64]=global_addr, [127:126]=2
//  group1: [17:16]=data_size(1 => 2B), [79:48]=tensor_dim0, [111:80]=tensor_dim1,
//          [127:112]=tile_dim0, [143:128]=tile_dim1, [159:144]=tile_dim2=0,
//          [207:160]=tensor_dim0_stride, [255:208]=tensor_dim1_stride(=0, 2D)
//  groups 2/3 (+ trailing group): zero (tile_dim3/4 unused -> up-to-2D tensor)
// All dims in data_size units. Tracked by TENSORcnt.
// This toolchain (clang-23 / therock-10.0 headers) uses the 6-arg builtin:
//   (uint32x4 g0, int32x8 g1, int32x4, int32x4, int32x8, i32 cpol)
// ---------------------------------------------------------------------------
static __device__ __forceinline__ void tdm_load_2d_f16(
    unsigned lds_byte_off, const _Float16* gptr,
    unsigned tensor_d0, unsigned tensor_d1, unsigned stride0_elems,
    unsigned tile_d0, unsigned tile_d1) {
  const unsigned long long ga = (unsigned long long)(size_t)gptr;
  v4u g0;
  g0[0] = 1u;                                     // count=1, user descriptor
  g0[1] = lds_byte_off;                           // lds_addr
  g0[2] = (unsigned)(ga & 0xFFFFFFFFu);           // global_addr[31:0]
  g0[3] = (unsigned)((ga >> 32) & 0x01FFFFFFu) |  // global_addr[56:32]
          (2u << 30);                             // type = 2 ("image")
  v8i g1;
  g1[0] = (int)(1u << 16);                        // data_size = 1 (2 bytes)
  g1[1] = (int)((tensor_d0 & 0xFFFFu) << 16);     // tensor_dim0[15:0] @ 63:48
  g1[2] = (int)((tensor_d0 >> 16) | ((tensor_d1 & 0xFFFFu) << 16));
  g1[3] = (int)((tensor_d1 >> 16) | (tile_d0 << 16));
  g1[4] = (int)tile_d1;                           // tile_dim1; tile_dim2 = 0
  g1[5] = (int)stride0_elems;                     // tensor_dim0_stride[31:0]
  g1[6] = 0;                                      // stride0[47:32], stride1[15:0]
  g1[7] = 0;                                      // stride1 unused (2-D)
  v4i z4 = {0, 0, 0, 0};
  v8i z8 = {0, 0, 0, 0, 0, 0, 0, 0};
  __builtin_amdgcn_tensor_load_to_lds(g0, g1, z4, z4, z8, 0);
}

// ---------------------------------------------------------------------------
// Cross-lane xor reductions within each 16-lane half (C-fragment rows span
// lanes 0-15 / 16-31) via ds_swizzle_b32 group-of-32 xor mode.
// ---------------------------------------------------------------------------
template <int XM>
static __device__ __forceinline__ float swz_xor(float v) {
  int x = __float_as_int(v);
  x = __builtin_amdgcn_ds_swizzle(x, (XM << 10) | 0x1F);
  return __int_as_float(x);
}
static __device__ __forceinline__ float red_max16(float v) {
  v = fmaxf(v, swz_xor<1>(v));
  v = fmaxf(v, swz_xor<2>(v));
  v = fmaxf(v, swz_xor<4>(v));
  v = fmaxf(v, swz_xor<8>(v));
  return v;
}
static __device__ __forceinline__ float red_sum16(float v) {
  v += swz_xor<1>(v);
  v += swz_xor<2>(v);
  v += swz_xor<4>(v);
  v += swz_xor<8>(v);
  return v;
}

// ---------------------------------------------------------------------------
// Fragment loaders from an LDS tile with row pitch `ld` halves.
// A fragment 16x32 (MxK): lane m = lane&15, half h = lane>>4,
//   element e: K = 16*(e>>3) + 8*h + (e&7)  -> two contiguous 8-elem chunks.
// B fragment 32x16 (KxN): lane n = lane&15, element e: K = 16*(e>>3) + 8*h + (e&7).
// ---------------------------------------------------------------------------
static __device__ __forceinline__ v16h frag_a_lds(const _Float16* tile, int ld,
                                                  int m0, int k0, int lane) {
  const int h = lane >> 4;
  const _Float16* p = tile + (m0 + (lane & 15)) * ld + k0 + 8 * h;
  v8h lo = *reinterpret_cast<const v8h*>(p);
  v8h hi = *reinterpret_cast<const v8h*>(p + 16);
  v16h a;
#pragma unroll
  for (int i = 0; i < 8; ++i) { a[i] = lo[i]; a[8 + i] = hi[i]; }
  return a;
}
static __device__ __forceinline__ v16h frag_b_lds(const _Float16* tile, int ld,
                                                  int n0, int k0, int lane) {
  const int h = lane >> 4;
  const _Float16* p = tile + (n0 + (lane & 15)) * ld + k0 + 8 * h;
  v8h lo = *reinterpret_cast<const v8h*>(p);
  v8h hi = *reinterpret_cast<const v8h*>(p + 16);
  v16h b;
#pragma unroll
  for (int i = 0; i < 8; ++i) { b[i] = lo[i]; b[8 + i] = hi[i]; }
  return b;
}

// ---------------------------------------------------------------------------
// One-time f32 -> f16 conversion (X and W copies in workspace).
// ---------------------------------------------------------------------------
__global__ __launch_bounds__(256) void cvt_kernel(const float* __restrict__ in,
                                                  _Float16* __restrict__ out,
                                                  int n8) {
  const int i = blockIdx.x * 256 + threadIdx.x;
  if (i < n8) {
    const float* p = in + (size_t)i * 8;
    v8h o;
#pragma unroll
    for (int j = 0; j < 8; ++j) o[j] = (_Float16)p[j];
    *reinterpret_cast<v8h*>(out + (size_t)i * 8) = o;
  }
}

// ---------------------------------------------------------------------------
// Projection GEMM: Out[m,n] = sum_k X[m,k] * W[n,k] + bias[n]   (all f16 in)
// A/B K-slabs staged into LDS by the Tensor Data Mover (wave 0 issues,
// double-buffered, s_wait_tensorcnt + one barrier per slab).
// LAYOUT 0: f16 row-major [M, D_MODEL]            (Q, K)
// LAYOUT 1: f16 transposed V: out[(b*D+n)*SEQ+s]  (V^T, packed v8h stores)
// LAYOUT 2: f32 row-major [M, D_MODEL]            (final output)
// Block 256 = 8 waves (4m x 2n), wave tile 32x64, WG tile 128x128.
// ---------------------------------------------------------------------------
template <int LAYOUT, typename TOut>
__global__ __launch_bounds__(256) void proj_kernel(
    const _Float16* __restrict__ X, const _Float16* __restrict__ W,
    const float* __restrict__ bias, TOut* __restrict__ out) {
  __shared__ __align__(32) _Float16 ldsA[2][128][32];  // [m][k] slab
  __shared__ __align__(32) _Float16 ldsB[2][128][32];  // [n][k] slab

  const int lane = threadIdx.x & 31;
  const int w    = threadIdx.x >> 5;
  const int wm   = w >> 1, wn = w & 1;
  const int m0   = blockIdx.x * 128;
  const int n0   = blockIdx.y * 128;
  const bool issuer = (w == 0);

  auto issue = [&](int buf, int k0) {
    tdm_load_2d_f16((unsigned)(size_t)&ldsA[buf][0][0],
                    X + (size_t)m0 * D_MODEL + k0,
                    D_MODEL, NB * SEQ, D_MODEL, 32, 128);
    tdm_load_2d_f16((unsigned)(size_t)&ldsB[buf][0][0],
                    W + (size_t)n0 * D_MODEL + k0,
                    D_MODEL, D_MODEL, D_MODEL, 32, 128);
  };

  if (issuer) issue(0, 0);

  v8f acc[2][4] = {};
  for (int k0 = 0, it = 0; k0 < D_MODEL; k0 += 32, ++it) {
    const int cur = it & 1;
    if (issuer) __builtin_amdgcn_s_wait_tensorcnt(0);  // tiles[cur] landed
    __syncthreads();  // all see tiles[cur]; all done reading tiles[cur^1]
    if (issuer && (k0 + 32 < D_MODEL)) issue(cur ^ 1, k0 + 32);

    const _Float16* At = &ldsA[cur][wm * 32][0];
    const _Float16* Bt = &ldsB[cur][wn * 64][0];
    v16h a[2], bf[4];
#pragma unroll
    for (int mt = 0; mt < 2; ++mt) a[mt] = frag_a_lds(At, 32, mt * 16, 0, lane);
#pragma unroll
    for (int nt = 0; nt < 4; ++nt) bf[nt] = frag_b_lds(Bt, 32, nt * 16, 0, lane);
#pragma unroll
    for (int mt = 0; mt < 2; ++mt)
#pragma unroll
      for (int nt = 0; nt < 4; ++nt)
        acc[mt][nt] = wmma16(a[mt], bf[nt], acc[mt][nt]);
  }

  // Epilogue: lane holds col n; VGPR r holds row m = base + r + 8*(lane>>4).
  const int l15 = lane & 15, h = lane >> 4;
#pragma unroll
  for (int nt = 0; nt < 4; ++nt) {
    const int n = n0 + wn * 64 + nt * 16 + l15;
    const float bvv = bias[n];
#pragma unroll
    for (int mt = 0; mt < 2; ++mt) {
      const int mb = m0 + wm * 32 + mt * 16 + 8 * h;
      if (LAYOUT == 1) {
        const int b = m0 >> 11;  // batch; 128-row tile never straddles
        v8h pk;
#pragma unroll
        for (int r = 0; r < 8; ++r) pk[r] = (_Float16)(acc[mt][nt][r] + bvv);
        *reinterpret_cast<v8h*>((_Float16*)out +
                                ((size_t)(b * D_MODEL + n)) * SEQ +
                                (mb - b * SEQ)) = pk;
      } else {
#pragma unroll
        for (int r = 0; r < 8; ++r)
          out[(size_t)(mb + r) * D_MODEL + n] = (TOut)(acc[mt][nt][r] + bvv);
      }
    }
  }
}

// ---------------------------------------------------------------------------
// Flash attention. Grid: (SEQ/64, NH, NB), block 128 = 4 waves.
// Wave owns 16 q rows; kv tiles of 32. K and V^T tiles are shared by all 4
// waves and staged via TDM (double-buffered). Softmax row stats live in
// registers; P is transposed to A-fragment layout via a wave-private LDS tile.
// ---------------------------------------------------------------------------
__global__ __launch_bounds__(128) void attn_kernel(
    const _Float16* __restrict__ Q, const _Float16* __restrict__ Kb,
    const _Float16* __restrict__ Vt, const int* __restrict__ amask,
    const int* __restrict__ kvflag_p, _Float16* __restrict__ ctx) {
  __shared__ __align__(32) _Float16 kt[2][32][64];     // [kv][d]
  __shared__ __align__(32) _Float16 vtile[2][64][32];  // [d][kv]
  __shared__ __align__(32) _Float16 plds[4][16][32];   // per-wave P bounce

  const int lane = threadIdx.x & 31;
  const int w    = threadIdx.x >> 5;
  const int l15  = lane & 15, h = lane >> 4;
  const int b = blockIdx.z, hh = blockIdx.y;
  const int q0 = blockIdx.x * 64 + w * 16;
  const int kvmode = *kvflag_p;
  const int* mrow = amask + b * SEQ;
  const bool issuer = (w == 0);

  auto issue = [&](int buf, int kv0) {
    tdm_load_2d_f16((unsigned)(size_t)&kt[buf][0][0],
                    Kb + (size_t)(b * SEQ + kv0) * D_MODEL + hh * HD,
                    D_MODEL, NB * SEQ, D_MODEL, HD, 32);
    tdm_load_2d_f16((unsigned)(size_t)&vtile[buf][0][0],
                    Vt + ((size_t)b * D_MODEL + hh * HD) * SEQ + kv0,
                    SEQ, NB * D_MODEL, SEQ, 32, HD);
  };

  // Q A-fragments for the two 32-wide head-dim chunks (loaded once).
  v16h aq[2];
  {
    const _Float16* p = Q + (size_t)(b * SEQ + q0 + l15) * D_MODEL + hh * HD;
#pragma unroll
    for (int kd = 0; kd < 2; ++kd) {
      v8h lo = *reinterpret_cast<const v8h*>(p + kd * 32 + 8 * h);
      v8h hi = *reinterpret_cast<const v8h*>(p + kd * 32 + 16 + 8 * h);
#pragma unroll
      for (int i = 0; i < 8; ++i) { aq[kd][i] = lo[i]; aq[kd][8 + i] = hi[i]; }
    }
  }

  if (issuer) issue(0, 0);

  v8f acc[4] = {};
  float rmax[8], rsum[8];
#pragma unroll
  for (int r = 0; r < 8; ++r) { rmax[r] = -3.0e38f; rsum[r] = 0.0f; }

  for (int kv0 = 0, it = 0; kv0 < SEQ; kv0 += 32, ++it) {
    const int cur = it & 1;
    if (issuer) __builtin_amdgcn_s_wait_tensorcnt(0);
    __syncthreads();
    if (issuer && (kv0 + 32 < SEQ)) issue(cur ^ 1, kv0 + 32);

    // --- scores: S[16x32] = Q(16x64) . K_tile^T, scale 1/sqrt(64) ---
    v8f sc[2] = {};
#pragma unroll
    for (int nt = 0; nt < 2; ++nt) {
#pragma unroll
      for (int kd = 0; kd < 2; ++kd) {
        v16h bk = frag_b_lds(&kt[cur][0][0], 64, nt * 16, kd * 32, lane);
        sc[nt] = wmma16(aq[kd], bk, sc[nt]);
      }
    }
    const int cm0 = mrow[kv0 + l15];
    const int cm1 = mrow[kv0 + 16 + l15];
#pragma unroll
    for (int r = 0; r < 8; ++r) {
      float v0 = sc[0][r] * 0.125f;
      float v1 = sc[1][r] * 0.125f;
      if (kvmode) {
        if (cm0 == 0) v0 += NEG_FILL;
        if (cm1 == 0) v1 += NEG_FILL;
      }
      sc[0][r] = v0; sc[1][r] = v1;
    }
    // --- online softmax ---
    float alpha[8];
#pragma unroll
    for (int r = 0; r < 8; ++r) {
      const float tm   = red_max16(fmaxf(sc[0][r], sc[1][r]));
      const float mnew = fmaxf(rmax[r], tm);
      alpha[r] = __expf(rmax[r] - mnew);
      rmax[r] = mnew;
      float p0 = __expf(sc[0][r] - mnew);
      float p1 = __expf(sc[1][r] - mnew);
      if (kvmode) {
        if (cm0 == 0) p0 = 0.0f;
        if (cm1 == 0) p1 = 0.0f;
      }
      sc[0][r] = p0; sc[1][r] = p1;
      rsum[r] = rsum[r] * alpha[r] + red_sum16(p0 + p1);
    }
#pragma unroll
    for (int nd = 0; nd < 4; ++nd)
#pragma unroll
      for (int r = 0; r < 8; ++r) acc[nd][r] *= alpha[r];

    // --- transpose P (C-layout) -> A-fragment via wave-private LDS tile ---
#pragma unroll
    for (int r = 0; r < 8; ++r) {
      plds[w][r + 8 * h][l15]      = (_Float16)sc[0][r];
      plds[w][r + 8 * h][16 + l15] = (_Float16)sc[1][r];
    }
    v16h ap = frag_a_lds(&plds[w][0][0], 32, 0, 0, lane);

    // --- acc += P(16x32) . V^T fragments (from staged tile) ---
#pragma unroll
    for (int nd = 0; nd < 4; ++nd) {
      v16h bv = frag_b_lds(&vtile[cur][0][0], 32, nd * 16, 0, lane);
      acc[nd] = wmma16(ap, bv, acc[nd]);
    }
  }

  // --- finalize: normalize, optional query-row mask, store context (f16) ---
  float inv[8], qsc[8];
#pragma unroll
  for (int r = 0; r < 8; ++r) {
    inv[r] = (rsum[r] > 0.0f) ? (1.0f / rsum[r]) : 0.0f;
    qsc[r] = 1.0f;
  }
  if (!kvmode) {
#pragma unroll
    for (int r = 0; r < 8; ++r) qsc[r] = (float)mrow[q0 + r + 8 * h];
  }
#pragma unroll
  for (int nd = 0; nd < 4; ++nd) {
    const int d = hh * HD + nd * 16 + l15;
#pragma unroll
    for (int r = 0; r < 8; ++r) {
      const int qr = q0 + r + 8 * h;
      ctx[(size_t)(b * SEQ + qr) * D_MODEL + d] =
          (_Float16)(acc[nd][r] * inv[r] * qsc[r]);
    }
  }
}

// ---------------------------------------------------------------------------
extern "C" void kernel_launch(void* const* d_in, const int* in_sizes, int n_in,
                              void* d_out, int out_size, void* d_ws,
                              size_t ws_size, hipStream_t stream) {
  (void)in_sizes; (void)n_in; (void)out_size; (void)ws_size;
  const float* q   = (const float*)d_in[0];
  const float* k   = (const float*)d_in[1];
  const float* v   = (const float*)d_in[2];
  const int*   am  = (const int*)d_in[3];
  const float* Wq  = (const float*)d_in[4];
  const float* bq  = (const float*)d_in[5];
  const float* Wk  = (const float*)d_in[6];
  const float* bk  = (const float*)d_in[7];
  const float* Wv  = (const float*)d_in[8];
  const float* bv  = (const float*)d_in[9];
  const float* Wp  = (const float*)d_in[10];
  const float* bp  = (const float*)d_in[11];
  const int* kvflag = (const int*)d_in[12];
  float* out = (float*)d_out;

  const size_t MN = (size_t)NB * SEQ * D_MODEL;   // 4M activations
  const size_t WN = (size_t)D_MODEL * D_MODEL;    // 1M weights
  _Float16* qbuf = (_Float16*)d_ws;
  _Float16* kbuf = qbuf + MN;
  _Float16* vt   = kbuf + MN;
  _Float16* ctx  = vt + MN;
  _Float16* q16  = ctx + MN;
  _Float16* k16  = q16 + MN;
  _Float16* v16  = k16 + MN;
  _Float16* wq16 = v16 + MN;
  _Float16* wk16 = wq16 + WN;
  _Float16* wv16 = wk16 + WN;
  _Float16* wp16 = wv16 + WN;

  // One-time f32 -> f16 conversions (keeps cvt out of the GEMM hot loops).
  const int actB = (int)(MN / 8 / 256), wB = (int)(WN / 8 / 256);
  cvt_kernel<<<actB, 256, 0, stream>>>(q, q16, (int)(MN / 8));
  cvt_kernel<<<actB, 256, 0, stream>>>(k, k16, (int)(MN / 8));
  cvt_kernel<<<actB, 256, 0, stream>>>(v, v16, (int)(MN / 8));
  cvt_kernel<<<wB, 256, 0, stream>>>(Wq, wq16, (int)(WN / 8));
  cvt_kernel<<<wB, 256, 0, stream>>>(Wk, wk16, (int)(WN / 8));
  cvt_kernel<<<wB, 256, 0, stream>>>(Wv, wv16, (int)(WN / 8));
  cvt_kernel<<<wB, 256, 0, stream>>>(Wp, wp16, (int)(WN / 8));

  dim3 gblk(256);
  dim3 ggrid((NB * SEQ) / 128, D_MODEL / 128);
  proj_kernel<0, _Float16><<<ggrid, gblk, 0, stream>>>(q16, wq16, bq, qbuf);
  proj_kernel<0, _Float16><<<ggrid, gblk, 0, stream>>>(k16, wk16, bk, kbuf);
  proj_kernel<1, _Float16><<<ggrid, gblk, 0, stream>>>(v16, wv16, bv, vt);

  dim3 ablk(128);
  dim3 agrid(SEQ / 64, NH, NB);
  attn_kernel<<<agrid, ablk, 0, stream>>>(qbuf, kbuf, vt, am, kvflag, ctx);

  proj_kernel<2, float><<<ggrid, gblk, 0, stream>>>(ctx, wp16, bp, out);
}
